// CrossAttention_2628519985163
// MI455X (gfx1250) — matrix operand order repaired and checked by reading, not compiled, not run on hardware
//
#include <hip/hip_runtime.h>
#include <hip/hip_bf16.h>

// ---------------------------------------------------------------------------
// Self-attention (B=2, N=4096, D=512, heads=8, dh=64) on gfx1250 (wave32).
// All GEMMs via v_wmma_f32_16x16x32_f16 (fp16 inputs, fp32 accumulate).
// Attention K/V tiles staged to LDS with async-to-LDS DMA (ASYNCcnt),
// double-buffered so DMA overlaps WMMA.
// ---------------------------------------------------------------------------

typedef __attribute__((ext_vector_type(16))) _Float16 v16h;
typedef __attribute__((ext_vector_type(8)))  _Float16 v8h;
typedef __attribute__((ext_vector_type(8)))  float    v8f;

#define WMMA_F32_F16(a, b, c) \
  __builtin_amdgcn_wmma_f32_16x16x32_f16(false, (a), false, (b), (short)0, (c), false, false)

// Problem constants
#define NTOK   4096
#define DMODEL 512
#define DHEAD  64
#define NHEADS 8
#define NBATCH 2
#define MTOT   (NBATCH * NTOK)   // 8192

// ---------------------------------------------------------------------------
// A-fragment (16x32 fp16, M x K) loader.
// Layout: lanes 0-15 hold M=lane, halves 0..15 = K 0..15
//         lanes 16-31: halves 0..15 = K 16..31 (matches contiguous B fill)
// => two 8-half (16B) runs at offsets g*16 and g*16+8 (g = lane/16).
// Works for both global and LDS pointers (generic addressing).
// ---------------------------------------------------------------------------
__device__ __forceinline__ v16h make_a_frag(const _Float16* row, int g) {
  v8h lo = *(const v8h*)(row + g * 16);
  v8h hi = *(const v8h*)(row + g * 16 + 8);
  v16h out;
#pragma unroll
  for (int i = 0; i < 8; ++i) { out[i] = lo[i]; out[i + 8] = hi[i]; }
  return out;
}

// Generic shared pointer -> 32-bit LDS offset (low 32 bits of the address).
__device__ __forceinline__ unsigned lds_off(const void* p) {
  return (unsigned)(size_t)p;
}

// Async DMA of 16 bytes global -> LDS (GLOBAL_LOAD_ASYNC_TO_LDS_B128, ASYNCcnt).
__device__ __forceinline__ void async_copy_b128(unsigned lds_dst, const void* gsrc) {
  unsigned long long ga = (unsigned long long)(size_t)gsrc;
  asm volatile("global_load_async_to_lds_b128 %0, %1, off"
               :: "v"(lds_dst), "v"(ga)
               : "memory");
}

__device__ __forceinline__ void wait_async0() {
  asm volatile("s_wait_asynccnt 0x0" ::: "memory");
}

// ---------------------------------------------------------------------------
// fp32 -> fp16 conversion of x
// ---------------------------------------------------------------------------
__global__ void cvt_x_kernel(const float* __restrict__ x, _Float16* __restrict__ x16, int n) {
  int i = blockIdx.x * 256 + threadIdx.x;
  if (i < n) x16[i] = (_Float16)x[i];
}

// ---------------------------------------------------------------------------
// Transpose + convert the four 512x512 weights: WT[n][k] = (fp16) W[k][n]
// ---------------------------------------------------------------------------
__global__ void cvt_wT_kernel(const float* __restrict__ Wq, const float* __restrict__ Wk,
                              const float* __restrict__ Wv, const float* __restrict__ Wo,
                              _Float16* __restrict__ WqT, _Float16* __restrict__ WkT,
                              _Float16* __restrict__ WvT, _Float16* __restrict__ WoT) {
  const float* W = (blockIdx.y == 0) ? Wq : (blockIdx.y == 1) ? Wk : (blockIdx.y == 2) ? Wv : Wo;
  _Float16*    T = (blockIdx.y == 0) ? WqT : (blockIdx.y == 1) ? WkT : (blockIdx.y == 2) ? WvT : WoT;
  int idx = blockIdx.x * 256 + threadIdx.x;       // 0 .. 512*512-1
  int k = idx >> 9;
  int n = idx & 511;
  T[n * DMODEL + k] = (_Float16)W[idx];
}

// ---------------------------------------------------------------------------
// QKV projection: proj = x16 @ W  (M=8192, K=512, Ncols=512)
// Each wave computes a 16x64 output strip (4 x 16x16 tiles, shared A frag).
// grid = (512, 2, 3), block = 128 (4 waves). z selects Wq/Wk/Wv.
// Q,K written head-major [bh][tok][64]; V written transposed [bh][64][tok].
// ---------------------------------------------------------------------------
__global__ __launch_bounds__(128) void proj_qkv_kernel(
    const _Float16* __restrict__ x16,
    const _Float16* __restrict__ WqT, const _Float16* __restrict__ WkT,
    const _Float16* __restrict__ WvT,
    _Float16* __restrict__ q16, _Float16* __restrict__ k16, _Float16* __restrict__ vt16) {
  const int lane = threadIdx.x & 31;
  const int g    = lane >> 4;
  const int ln   = lane & 15;
  const int wave = threadIdx.x >> 5;
  const int row0 = blockIdx.x * 16;                       // token-row tile base
  const int col0 = (blockIdx.y * 4 + wave) * 64;          // 64-wide column strip
  const _Float16* WT = (blockIdx.z == 0) ? WqT : (blockIdx.z == 1) ? WkT : WvT;

  v8f acc[4];
#pragma unroll
  for (int t = 0; t < 4; ++t) acc[t] = {};

  const _Float16* arow = x16 + (size_t)(row0 + ln) * DMODEL;
#pragma unroll 4
  for (int kk = 0; kk < DMODEL; kk += 32) {
    v16h a = make_a_frag(arow + kk, g);
#pragma unroll
    for (int t = 0; t < 4; ++t) {
      v16h b = *(const v16h*)(WT + (size_t)(col0 + t * 16 + ln) * DMODEL + kk + g * 16);
      acc[t] = WMMA_F32_F16(a, b, acc[t]);
    }
  }

  const int bidx = row0 >> 12;          // batch (tile never straddles: 4096 % 16 == 0)
  const int i0   = row0 & (NTOK - 1);
  const int h    = col0 >> 6;           // head (strip never straddles: 64-aligned)
  const size_t bh = (size_t)(bidx * NHEADS + h);

  if (blockIdx.z < 2) {
    _Float16* dst = (blockIdx.z == 0) ? q16 : k16;
#pragma unroll
    for (int t = 0; t < 4; ++t) {
      int d = t * 16 + ln;
#pragma unroll
      for (int r = 0; r < 8; ++r) {
        int i = i0 + r + 8 * g;
        dst[(bh * NTOK + i) * DHEAD + d] = (_Float16)acc[t][r];
      }
    }
  } else {
#pragma unroll
    for (int t = 0; t < 4; ++t) {
      int d = t * 16 + ln;
#pragma unroll
      for (int r = 0; r < 8; ++r) {
        int i = i0 + r + 8 * g;
        vt16[(bh * DHEAD + d) * NTOK + i] = (_Float16)acc[t][r];  // transposed
      }
    }
  }
}

// ---------------------------------------------------------------------------
// Flash attention: per (b,h), 64 Q rows per block (16 per wave).
// K/V 64-wide tiles staged to LDS via async-to-LDS DMA, double buffered so
// the DMA of tile j+1 overlaps the WMMA work on tile j. Online softmax.
// grid = (4096/64, 16), block = 128 (4 waves).
// ---------------------------------------------------------------------------
__global__ __launch_bounds__(128) void flash_attn_kernel(
    const _Float16* __restrict__ q16, const _Float16* __restrict__ k16,
    const _Float16* __restrict__ vt16, _Float16* __restrict__ o16) {
  __shared__ alignas(16) _Float16 shK[2][64 * 64];   // K tile [tok][d]
  __shared__ alignas(16) _Float16 shV[2][64 * 64];   // V tile [d][tok]
  __shared__ alignas(32) _Float16 shP[4][16 * 72];   // per-wave P tile, padded rows

  const int tid  = threadIdx.x;
  const int lane = tid & 31;
  const int g    = lane >> 4;
  const int ln   = lane & 15;
  const int wave = tid >> 5;
  const int bh   = blockIdx.y;                       // 0..15
  const int qrow = blockIdx.x * 64 + wave * 16;      // token index within (b,h)
  const float scale = 0.125f;                        // dh^-0.5 = 1/8

  const _Float16* kb = k16 + (size_t)bh * NTOK * DHEAD;
  const _Float16* vb = vt16 + (size_t)bh * DHEAD * NTOK;

  // Q fragments for this wave's 16 rows (held for the whole pass)
  const _Float16* qbase = q16 + ((size_t)bh * NTOK + qrow + ln) * DHEAD;
  const v16h aq0 = make_a_frag(qbase + 0, g);
  const v16h aq1 = make_a_frag(qbase + 32, g);

  v8f accO[4];
#pragma unroll
  for (int t = 0; t < 4; ++t) accO[t] = {};
  float m[8], l[8];
#pragma unroll
  for (int r = 0; r < 8; ++r) { m[r] = -3.0e38f; l[r] = 0.0f; }

  _Float16* P = &shP[wave][0];

  // Stage tile j0 (64 keys) into LDS buffer `buf`: all 128 threads cooperate.
  auto stage = [&](int buf, int j0) {
    // K tile: rows j0..j0+63, contiguous 4096 halves (8 KB) in memory.
    const _Float16* kg = kb + (size_t)j0 * DHEAD;
#pragma unroll
    for (int rr = 0; rr < 4; ++rr) {
      int c = tid + rr * 128;                        // 16B chunk id, 0..511
      async_copy_b128(lds_off(&shK[buf][c * 8]), kg + c * 8);
    }
    // V tile: 64 rows (d) x 128 B, row stride NTOK halves in VT.
#pragma unroll
    for (int rr = 0; rr < 4; ++rr) {
      int c = tid + rr * 128;
      int row = c >> 3, col8 = (c & 7) * 8;
      async_copy_b128(lds_off(&shV[buf][row * 64 + col8]),
                      vb + (size_t)row * NTOK + j0 + col8);
    }
  };

  stage(0, 0);   // prefetch first tile

  for (int jt = 0; jt < NTOK / 64; ++jt) {
    const int cur = jt & 1;
    wait_async0();            // my async DMAs (incl. prefetch) are in LDS
    __syncthreads();          // everyone's chunks visible; buf[cur] ready
    if (jt + 1 < NTOK / 64) stage(1 - cur, (jt + 1) * 64);  // overlap next DMA

    const _Float16* kt = &shK[cur][0];
    const _Float16* vt = &shV[cur][0];

    // ---- S = scale * Q K^T  (4 x 16x16 tiles over 64 key columns) ----
    v8f s[4];
#pragma unroll
    for (int t = 0; t < 4; ++t) {
      const _Float16* krow = kt + (t * 16 + ln) * DHEAD + g * 16;
      v16h b0 = *(const v16h*)(krow);
      v16h b1 = *(const v16h*)(krow + 32);
      v8f c = {};
      c = WMMA_F32_F16(aq0, b0, c);
      c = WMMA_F32_F16(aq1, b1, c);
      s[t] = c;
    }
#pragma unroll
    for (int t = 0; t < 4; ++t)
#pragma unroll
      for (int r = 0; r < 8; ++r) s[t][r] *= scale;

    // ---- row max across the 64 columns (intra-lane then 16-lane butterfly) ----
    float mt[8];
#pragma unroll
    for (int r = 0; r < 8; ++r) {
      float v = s[0][r];
      v = fmaxf(v, s[1][r]); v = fmaxf(v, s[2][r]); v = fmaxf(v, s[3][r]);
      mt[r] = v;
    }
#pragma unroll
    for (int mask = 1; mask <= 8; mask <<= 1)
#pragma unroll
      for (int r = 0; r < 8; ++r) mt[r] = fmaxf(mt[r], __shfl_xor(mt[r], mask));

    // ---- online-softmax rescale ----
    float alpha[8];
#pragma unroll
    for (int r = 0; r < 8; ++r) {
      float mn = fmaxf(m[r], mt[r]);
      alpha[r] = __expf(m[r] - mn);
      m[r] = mn;
      l[r] *= alpha[r];
    }
#pragma unroll
    for (int t = 0; t < 4; ++t)
#pragma unroll
      for (int r = 0; r < 8; ++r) accO[t][r] *= alpha[r];

    // ---- P = exp(S - m): accumulate row-sum, spill to LDS as fp16 ----
    float rs[8];
#pragma unroll
    for (int r = 0; r < 8; ++r) rs[r] = 0.0f;
#pragma unroll
    for (int t = 0; t < 4; ++t) {
      int col = t * 16 + ln;
#pragma unroll
      for (int r = 0; r < 8; ++r) {
        float p = __expf(s[t][r] - m[r]);
        rs[r] += p;
        P[(r + 8 * g) * 72 + col] = (_Float16)p;
      }
    }
#pragma unroll
    for (int mask = 1; mask <= 8; mask <<= 1)
#pragma unroll
      for (int r = 0; r < 8; ++r) rs[r] += __shfl_xor(rs[r], mask);
#pragma unroll
    for (int r = 0; r < 8; ++r) l[r] += rs[r];

    // ---- O += P @ V   (same-wave DS ops are in-order: no barrier needed) ----
#pragma unroll
    for (int kk = 0; kk < 64; kk += 32) {
      v16h ap = make_a_frag(P + ln * 72 + kk, g);
#pragma unroll
      for (int t = 0; t < 4; ++t) {
        v16h bv = *(const v16h*)(vt + (t * 16 + ln) * DHEAD + kk + g * 16);
        accO[t] = WMMA_F32_F16(ap, bv, accO[t]);
      }
    }
    __syncthreads();          // all reads of buf[cur] done before it is restaged
  }

  // ---- normalize and write O as fp16 [b*N][h*64+d] for the output GEMM ----
  float inv[8];
#pragma unroll
  for (int r = 0; r < 8; ++r) inv[r] = 1.0f / fmaxf(l[r], 1e-30f);

  const int bidx = bh >> 3;
  const int h    = bh & 7;
#pragma unroll
  for (int t = 0; t < 4; ++t) {
    int d = t * 16 + ln;
#pragma unroll
    for (int r = 0; r < 8; ++r) {
      int i = qrow + r + 8 * g;
      o16[((size_t)bidx * NTOK + i) * DMODEL + h * DHEAD + d] = (_Float16)(accO[t][r] * inv[r]);
    }
  }
}

// ---------------------------------------------------------------------------
// Output projection: out = O16 @ Wo + bo, fp32 result.
// grid = (512, 2), block = 128 (4 waves), 16x64 strip per wave.
// ---------------------------------------------------------------------------
__global__ __launch_bounds__(128) void out_proj_kernel(
    const _Float16* __restrict__ o16, const _Float16* __restrict__ WoT,
    const float* __restrict__ bo, float* __restrict__ out) {
  const int lane = threadIdx.x & 31;
  const int g    = lane >> 4;
  const int ln   = lane & 15;
  const int wave = threadIdx.x >> 5;
  const int row0 = blockIdx.x * 16;
  const int col0 = (blockIdx.y * 4 + wave) * 64;

  v8f acc[4];
#pragma unroll
  for (int t = 0; t < 4; ++t) acc[t] = {};

  const _Float16* arow = o16 + (size_t)(row0 + ln) * DMODEL;
#pragma unroll 4
  for (int kk = 0; kk < DMODEL; kk += 32) {
    v16h a = make_a_frag(arow + kk, g);
#pragma unroll
    for (int t = 0; t < 4; ++t) {
      v16h b = *(const v16h*)(WoT + (size_t)(col0 + t * 16 + ln) * DMODEL + kk + g * 16);
      acc[t] = WMMA_F32_F16(a, b, acc[t]);
    }
  }

#pragma unroll
  for (int t = 0; t < 4; ++t) {
    int col = col0 + t * 16 + ln;
    float bias = bo[col];
#pragma unroll
    for (int r = 0; r < 8; ++r) {
      out[(size_t)(row0 + r + 8 * g) * DMODEL + col] = acc[t][r] + bias;
    }
  }
}

// ---------------------------------------------------------------------------
// Launch
// ---------------------------------------------------------------------------
extern "C" void kernel_launch(void* const* d_in, const int* in_sizes, int n_in,
                              void* d_out, int out_size, void* d_ws, size_t ws_size,
                              hipStream_t stream) {
  const float* x  = (const float*)d_in[0];
  const float* Wq = (const float*)d_in[1];
  const float* Wk = (const float*)d_in[2];
  const float* Wv = (const float*)d_in[3];
  const float* Wo = (const float*)d_in[4];
  const float* bo = (const float*)d_in[5];
  float* out = (float*)d_out;

  char* ws = (char*)d_ws;
  const size_t SZ_X16 = (size_t)MTOT * DMODEL * sizeof(_Float16);   // 8 MB
  const size_t SZ_W   = (size_t)DMODEL * DMODEL * sizeof(_Float16); // 512 KB
  _Float16* x16  = (_Float16*)(ws);
  _Float16* WqT  = (_Float16*)(ws + SZ_X16);
  _Float16* WkT  = (_Float16*)(ws + SZ_X16 + SZ_W);
  _Float16* WvT  = (_Float16*)(ws + SZ_X16 + 2 * SZ_W);
  _Float16* WoT  = (_Float16*)(ws + SZ_X16 + 3 * SZ_W);
  _Float16* q16  = (_Float16*)(ws + SZ_X16 + 4 * SZ_W);
  _Float16* k16  = (_Float16*)(ws + 2 * SZ_X16 + 4 * SZ_W);
  _Float16* vt16 = (_Float16*)(ws + 3 * SZ_X16 + 4 * SZ_W);
  _Float16* o16  = (_Float16*)(ws + 4 * SZ_X16 + 4 * SZ_W);
  // total footprint: 5*8MB + 2MB = 42 MB

  // 1) convert inputs to fp16 (+ weight transpose)
  cvt_x_kernel<<<(MTOT * DMODEL) / 256, 256, 0, stream>>>(x, x16, MTOT * DMODEL);
  cvt_wT_kernel<<<dim3((DMODEL * DMODEL) / 256, 4), 256, 0, stream>>>(
      Wq, Wk, Wv, Wo, WqT, WkT, WvT, WoT);

  // 2) Q/K/V projections (V stored transposed per head)
  proj_qkv_kernel<<<dim3(MTOT / 16, 2, 3), 128, 0, stream>>>(
      x16, WqT, WkT, WvT, q16, k16, vt16);

  // 3) flash attention per (batch, head), async-LDS staged K/V
  flash_attn_kernel<<<dim3(NTOK / 64, NBATCH * NHEADS), 128, 0, stream>>>(
      q16, k16, vt16, o16);

  // 4) output projection + bias (fp32 out)
  out_proj_kernel<<<dim3(MTOT / 16, 2), 128, 0, stream>>>(o16, WoT, bo, out);
}
